// CausalSelfAttention_56856777064504
// MI455X (gfx1250) — matrix operand order repaired and checked by reading, not compiled, not run on hardware
//
#include <hip/hip_runtime.h>
#include <math.h>

// ---------------------------------------------------------------------------
// CDNA5 (gfx1250) causal self-attention: bf16 WMMA pipeline
//   B=2, L=2048, D=1024, H=16, hd=64
// - Weights transposed ([N,K] bf16), V transposed ([B,H,hd,L] bf16):
//   every WMMA fragment loads as b128.
// - Flash softmax: row-max via stage-major batched ds_bpermute butterflies
//   (4 dscnt waits/block instead of 32); row-sum via an extra WMMA vs ones.
// - Causal mask applied only on the single diagonal-straddling block.
// - global_prefetch_b8 stream-ahead on weight rows and next K/V block.
// ---------------------------------------------------------------------------

typedef __attribute__((ext_vector_type(8)))  __bf16 v8bf;
typedef __attribute__((ext_vector_type(16))) __bf16 v16bf;
typedef __attribute__((ext_vector_type(8)))  float  v8f;

static constexpr int B_  = 2;
static constexpr int L_  = 2048;
static constexpr int D_  = 1024;
static constexpr int H_  = 16;
static constexpr int HD_ = 64;   // head dim
static constexpr int TD_ = 3 * D_;

__device__ __forceinline__ v16bf cat8(v8bf lo, v8bf hi) {
  return __builtin_shufflevector(lo, hi, 0, 1, 2, 3, 4, 5, 6, 7,
                                 8, 9, 10, 11, 12, 13, 14, 15);
}
__device__ __forceinline__ v8bf ld8(const __bf16* p) {
  return *(const v8bf*)p;     // 16-byte aligned by construction
}

// ---------------------------------------------------------------------------
// conversions
// ---------------------------------------------------------------------------
__global__ void k_f32_to_bf16(const float* __restrict__ in,
                              __bf16* __restrict__ out, int n) {
  int i = blockIdx.x * blockDim.x + threadIdx.x;
  int stride = gridDim.x * blockDim.x;
  for (; i < n; i += stride) out[i] = (__bf16)in[i];
}

// fp32 [K,N] -> bf16 [N,K]
__global__ void k_f32_to_bf16_T(const float* __restrict__ in,
                                __bf16* __restrict__ out, int K, int N) {
  int i = blockIdx.x * blockDim.x + threadIdx.x;
  int stride = gridDim.x * blockDim.x;
  const int total = K * N;
  for (; i < total; i += stride) {
    const int k = i % K;
    const int n = i / K;
    out[i] = (__bf16)in[(size_t)k * N + n];
  }
}

// vt[((b*H + h)*hd + d)*L + l] = qkv[b, l, 2D + h*hd + d]
__global__ void k_build_vt(const __bf16* __restrict__ qkv,
                           __bf16* __restrict__ vt) {
  int i = blockIdx.x * blockDim.x + threadIdx.x;
  int stride = gridDim.x * blockDim.x;
  const int total = B_ * H_ * HD_ * L_;
  for (; i < total; i += stride) {
    const int l = i % L_;
    const int d = (i / L_) % HD_;
    const int h = (i / (L_ * HD_)) % H_;
    const int b = i / (L_ * HD_ * H_);
    vt[i] = qkv[((size_t)b * L_ + l) * TD_ + 2 * D_ + h * HD_ + d];
  }
}

// ---------------------------------------------------------------------------
// GEMM: C[M,N] = A[M,K](bf16) * Bt[N,K](bf16, transposed) + bias[N](f32)
// One wave: 16(M) x 64(N) tile, 4 accumulators, A-frag reused 4x.
// ---------------------------------------------------------------------------
template <typename OutT>
__global__ void k_gemm_bias(const __bf16* __restrict__ A,
                            const __bf16* __restrict__ Bt,
                            const float*  __restrict__ bias,
                            OutT* __restrict__ C,
                            int M, int N, int K) {
  const int lane  = threadIdx.x & 31;
  const int wid   = blockIdx.x * (blockDim.x >> 5) + (threadIdx.x >> 5);
  const int ntN   = N >> 6;
  const int total = (M >> 4) * ntN;
  if (wid >= total) return;

  const int m0 = (wid / ntN) << 4;
  const int n0 = (wid % ntN) << 6;

  const int col   = lane & 15;
  const int khalf = lane >> 4;

  const __bf16* ap = A + (size_t)(m0 + col) * K;
  const __bf16* bp[4];
#pragma unroll
  for (int t = 0; t < 4; ++t)
    bp[t] = Bt + (size_t)(n0 + t * 16 + col) * K;

  v8f acc[4] = {v8f{}, v8f{}, v8f{}, v8f{}};

  for (int kk = 0; kk < K; kk += 32) {
    if (kk + 64 < K) {
      // stream-ahead hint for the weight rows (global_prefetch_b8)
#pragma unroll
      for (int t = 0; t < 4; ++t)
        __builtin_prefetch(bp[t] + kk + 64, 0, 3);
    }
    const v16bf a = cat8(ld8(ap + kk + khalf * 16),
                         ld8(ap + kk + khalf * 16 + 8));
#pragma unroll
    for (int t = 0; t < 4; ++t) {
      const __bf16* p = bp[t] + kk + khalf * 16;
      const v16bf b = cat8(ld8(p), ld8(p + 8));
      acc[t] = __builtin_amdgcn_wmma_f32_16x16x32_bf16(
          false, a, false, b, (short)0, acc[t], false, false);
    }
  }

  const int rbase = m0 + (khalf << 3);
#pragma unroll
  for (int t = 0; t < 4; ++t) {
    const int ccol = n0 + t * 16 + col;
    const float bv = bias[ccol];
#pragma unroll
    for (int r = 0; r < 8; ++r)
      C[(size_t)(rbase + r) * N + ccol] = (OutT)(acc[t][r] + bv);
  }
}

// ---------------------------------------------------------------------------
// flash-attention 32-key block (MASKED only for the diagonal block)
// ---------------------------------------------------------------------------
template <bool MASKED>
__device__ __forceinline__ void attn_block(
    int j0, int q0, int col, int khalf, int rbase,
    const __bf16* __restrict__ Kb, const __bf16* __restrict__ Vt,
    const v16bf (&qa)[2], const v16bf& ones,
    float (&m_i)[8], float (&l_i)[8], v8f (&Oacc)[4],
    __bf16 (*Pl)[32]) {
  const float NEG_INF = -__builtin_inff();
  const float scale = 0.125f;            // 1/sqrt(64)

  // prefetch next key block's K rows and V^T runs
  if (j0 + 32 + 31 < L_) {
#pragma unroll
    for (int st = 0; st < 2; ++st)
      __builtin_prefetch(Kb + (size_t)(j0 + 32 + st * 16 + col) * TD_, 0, 3);
#pragma unroll
    for (int t = 0; t < 4; ++t)
      __builtin_prefetch(Vt + (size_t)(t * 16 + col) * L_ + j0 + 32, 0, 3);
  }

  // S = Q * K^T : two 16x16 score tiles (keys j0..+15, j0+16..+31)
  v8f s[2] = {v8f{}, v8f{}};
#pragma unroll
  for (int st = 0; st < 2; ++st) {
    const __bf16* kp = Kb + (size_t)(j0 + st * 16 + col) * TD_;
#pragma unroll
    for (int half = 0; half < 2; ++half) {
      const __bf16* p = kp + half * 32 + khalf * 16;
      const v16bf kf = cat8(ld8(p), ld8(p + 8));
      s[st] = __builtin_amdgcn_wmma_f32_16x16x32_bf16(
          false, qa[half], false, kf, (short)0, s[st], false, false);
    }
  }

  // scale + (mask) + per-lane row max
  float mx[8];
#pragma unroll
  for (int r = 0; r < 8; ++r) {
    float v0 = s[0][r] * scale;
    float v1 = s[1][r] * scale;
    if (MASKED) {
      const int qrow = q0 + rbase + r;
      v0 = ((j0 + col)      > qrow) ? NEG_INF : v0;
      v1 = ((j0 + 16 + col) > qrow) ? NEG_INF : v1;
    }
    s[0][r] = v0;
    s[1][r] = v1;
    mx[r] = fmaxf(v0, v1);
  }

  // stage-major butterfly over the 16-lane row groups: 8 independent trees,
  // all permutes of a stage issued back-to-back (one dscnt wait per stage)
#pragma unroll
  for (int off = 1; off < 16; off <<= 1) {
    float tsh[8];
#pragma unroll
    for (int r = 0; r < 8; ++r) tsh[r] = __shfl_xor(mx[r], off, 32);
#pragma unroll
    for (int r = 0; r < 8; ++r) mx[r] = fmaxf(mx[r], tsh[r]);
  }

  __syncthreads();   // protect Pl WAR from previous block

  // rescale running state, exponentiate, stage P into LDS
#pragma unroll
  for (int r = 0; r < 8; ++r) {
    const float mnew  = fmaxf(m_i[r], mx[r]);
    const float alpha = __expf(m_i[r] - mnew);
    m_i[r] = mnew;
    l_i[r] *= alpha;
#pragma unroll
    for (int t = 0; t < 4; ++t) Oacc[t][r] *= alpha;

    Pl[rbase + r][col]      = (__bf16)__expf(s[0][r] - mnew);
    Pl[rbase + r][16 + col] = (__bf16)__expf(s[1][r] - mnew);
  }

  __syncthreads();

  // P as 16x32 A-fragment from LDS (two ds_load_b128)
  const __bf16* pp = &Pl[col][khalf * 16];
  const v16bf pa = cat8(ld8(pp), ld8(pp + 8));

  // row-sum of P via matrix unit: P x Ones -> every lane gets its rows' sums
  v8f rsum = __builtin_amdgcn_wmma_f32_16x16x32_bf16(
      false, pa, false, ones, (short)0, v8f{}, false, false);
#pragma unroll
  for (int r = 0; r < 8; ++r) l_i[r] += rsum[r];

  // O += P * V  (V^T gives contiguous 32B key-runs per d column)
#pragma unroll
  for (int t = 0; t < 4; ++t) {
    const __bf16* vp = Vt + (size_t)(t * 16 + col) * L_ + j0 + khalf * 16;
    const v16bf vf = cat8(ld8(vp), ld8(vp + 8));
    Oacc[t] = __builtin_amdgcn_wmma_f32_16x16x32_bf16(
        false, pa, false, vf, (short)0, Oacc[t], false, false);
  }
}

// ---------------------------------------------------------------------------
// Flash attention (causal), one wave (32 threads) per (b, h, 16-query tile).
// ---------------------------------------------------------------------------
__global__ void k_flash_attn(const __bf16* __restrict__ qkv,
                             const __bf16* __restrict__ vt,
                             __bf16* __restrict__ out) {
  __shared__ __bf16 Pl[16][32];

  const int lane  = threadIdx.x & 31;
  const int col   = lane & 15;
  const int khalf = lane >> 4;

  const int nqt = L_ >> 4;
  const int qt  = blockIdx.x % nqt;
  const int h   = (blockIdx.x / nqt) % H_;
  const int b   = blockIdx.x / (nqt * H_);
  const int q0  = qt << 4;

  const size_t base = (size_t)b * L_ * TD_ + (size_t)h * HD_;
  const __bf16* Qb = qkv + base;
  const __bf16* Kb = qkv + base + D_;
  const __bf16* Vt = vt + (size_t)(b * H_ + h) * HD_ * L_;

  // Q as two 16x32 A-fragments (d = 0..31, 32..63)
  const __bf16* qp = Qb + (size_t)(q0 + col) * TD_;
  v16bf qa[2];
#pragma unroll
  for (int half = 0; half < 2; ++half) {
    const __bf16* p = qp + half * 32 + khalf * 16;
    qa[half] = cat8(ld8(p), ld8(p + 8));
  }

  v16bf ones;
#pragma unroll
  for (int i = 0; i < 16; ++i) ones[i] = (__bf16)1.0f;

  float m_i[8], l_i[8];
#pragma unroll
  for (int r = 0; r < 8; ++r) { m_i[r] = -__builtin_inff(); l_i[r] = 0.0f; }
  v8f Oacc[4] = {v8f{}, v8f{}, v8f{}, v8f{}};

  const int rbase = khalf << 3;
  const int jbmax = (q0 + 15) >> 5;   // blocks 0..jbmax-1 fully unmasked

  for (int jb = 0; jb < jbmax; ++jb)
    attn_block<false>(jb << 5, q0, col, khalf, rbase, Kb, Vt,
                      qa, ones, m_i, l_i, Oacc, Pl);
  attn_block<true>(jbmax << 5, q0, col, khalf, rbase, Kb, Vt,
                   qa, ones, m_i, l_i, Oacc, Pl);

  // normalize and store: out[b, q0+row, h*64 + d]  (bf16)
#pragma unroll
  for (int r = 0; r < 8; ++r) {
    const float inv = 1.0f / l_i[r];
    const int row = q0 + rbase + r;
#pragma unroll
    for (int t = 0; t < 4; ++t) {
      out[(size_t)b * L_ * D_ + (size_t)row * D_ + h * HD_ + t * 16 + col] =
          (__bf16)(Oacc[t][r] * inv);
    }
  }
}

// ---------------------------------------------------------------------------
// launch
// ---------------------------------------------------------------------------
extern "C" void kernel_launch(void* const* d_in, const int* in_sizes, int n_in,
                              void* d_out, int out_size, void* d_ws, size_t ws_size,
                              hipStream_t stream) {
  const float* x     = (const float*)d_in[0];   // [B,L,D]
  const float* W_qkv = (const float*)d_in[1];   // [D,3D]
  const float* b_qkv = (const float*)d_in[2];   // [3D]
  const float* W_out = (const float*)d_in[3];   // [D,D]
  const float* b_out = (const float*)d_in[4];   // [D]
  float* outp = (float*)d_out;                  // [B,L,D]

  const int M = B_ * L_;                        // 4096

  char* ws = (char*)d_ws;
  __bf16* x_bf     = (__bf16*)ws;  ws += (size_t)M * D_ * 2;     // 8 MiB
  __bf16* WqkvT_bf = (__bf16*)ws;  ws += (size_t)D_ * TD_ * 2;   // 6 MiB  [3D, D]
  __bf16* WoutT_bf = (__bf16*)ws;  ws += (size_t)D_ * D_ * 2;    // 2 MiB  [D, D]
  __bf16* qkv_bf   = (__bf16*)ws;  ws += (size_t)M * TD_ * 2;    // 24 MiB
  __bf16* attn_bf  = (__bf16*)ws;  ws += (size_t)M * D_ * 2;     // 8 MiB
  __bf16* vt_bf    = (__bf16*)ws;                                // 8 MiB

  k_f32_to_bf16<<<2048, 256, 0, stream>>>(x, x_bf, M * D_);
  k_f32_to_bf16_T<<<2048, 256, 0, stream>>>(W_qkv, WqkvT_bf, D_, TD_);
  k_f32_to_bf16_T<<<1024, 256, 0, stream>>>(W_out, WoutT_bf, D_, D_);

  {
    const int waves = (M >> 4) * (TD_ >> 6);      // 256 * 48
    k_gemm_bias<__bf16><<<(waves + 3) / 4, 128, 0, stream>>>(
        x_bf, WqkvT_bf, b_qkv, qkv_bf, M, TD_, D_);
  }

  k_build_vt<<<2048, 256, 0, stream>>>(qkv_bf, vt_bf);
  k_flash_attn<<<B_ * H_ * (L_ >> 4), 32, 0, stream>>>(qkv_bf, vt_bf, attn_bf);

  {
    const int waves = (M >> 4) * (D_ >> 6);       // 256 * 16
    k_gemm_bias<float><<<(waves + 3) / 4, 128, 0, stream>>>(
        attn_bf, WoutT_bf, b_out, outp, M, D_, D_);
  }
  (void)in_sizes; (void)n_in; (void)out_size; (void)ws_size;
}
